// Attention_60215441490609
// MI455X (gfx1250) — matrix operand order repaired and checked by reading, not compiled, not run on hardware
//
#include <hip/hip_runtime.h>

typedef __attribute__((ext_vector_type(16))) __bf16 v16bf;
typedef __attribute__((ext_vector_type(8)))  __bf16 v8bf;
typedef __attribute__((ext_vector_type(4)))  __bf16 v4bf;
typedef __attribute__((ext_vector_type(8)))  float  v8f;

#define B_  2
#define S_  2048
#define D_  1024
#define H_  16
#define DH_ 64

// ---------------------------------------------------------------------------
// WMMA helpers (CDNA5 16x16x32 bf16, fp32 accumulate)
// ---------------------------------------------------------------------------
__device__ __forceinline__ v8f wmma_bf16(v16bf a, v16bf b, v8f c) {
  return __builtin_amdgcn_wmma_f32_16x16x32_bf16(
      /*neg_a=*/false, a, /*neg_b=*/false, b,
      /*c_mod=*/(short)0, c, /*reuse_a=*/false, /*reuse_b=*/false);
}

__device__ __forceinline__ v16bf combine16(const __bf16* p0, const __bf16* p1) {
  v8bf lo = *(const v8bf*)p0;
  v8bf hi = *(const v8bf*)p1;
  v16bf r;
#pragma unroll
  for (int i = 0; i < 8; ++i) { r[i] = lo[i]; r[i + 8] = hi[i]; }
  return r;
}

// A fragment (16x32 bf16) from row-major [rows][stride] tile.
// lane: m = l%16; VGPR0-3: k = half*8+0..7 ; VGPR4-7: k = 16+half*8+0..7
__device__ __forceinline__ v16bf load_a_frag(const __bf16* p, int row, int stride,
                                             int kk, int half_) {
  const __bf16* b = p + (size_t)row * stride + kk + half_ * 8;
  return combine16(b, b + 16);
}

// B fragment (32x16 bf16) from k-contiguous layout [n][stride] (n = B column).
// lane: n = l%16; VGPR0-3: k = half*8+0..7 ; VGPR4-7: k = 16+half*8+0..7
__device__ __forceinline__ v16bf load_b_frag(const __bf16* p, int row, int stride,
                                             int kk, int half_) {
  const __bf16* b = p + (size_t)row * stride + kk + half_ * 8;
  return combine16(b, b + 16);
}

// ---------------------------------------------------------------------------
// Pre-conversion kernels
// ---------------------------------------------------------------------------
__global__ void f32_to_bf16_kernel(const float* __restrict__ src,
                                   __bf16* __restrict__ dst, int n4) {
  int i = blockIdx.x * blockDim.x + threadIdx.x;
  if (i < n4) {
    float4 v = ((const float4*)src)[i];
    v4bf o;
    o[0] = (__bf16)v.x; o[1] = (__bf16)v.y; o[2] = (__bf16)v.z; o[3] = (__bf16)v.w;
    ((v4bf*)dst)[i] = o;
  }
}

// dst[c][r] = (bf16) src[r][c]   (R, C multiples of 32)
__global__ void transpose_to_bf16_kernel(const float* __restrict__ src,
                                         __bf16* __restrict__ dst, int R, int C) {
  __shared__ float tile[32][33];
  int c0 = blockIdx.x * 32, r0 = blockIdx.y * 32;
  int tx = threadIdx.x, ty = threadIdx.y;
#pragma unroll
  for (int i = ty; i < 32; i += 8)
    tile[i][tx] = src[(size_t)(r0 + i) * C + c0 + tx];
  __syncthreads();
#pragma unroll
  for (int i = ty; i < 32; i += 8)
    dst[(size_t)(c0 + i) * R + r0 + tx] = (__bf16)tile[tx][i];
}

// ---------------------------------------------------------------------------
// Stage 1: QKV = x @ w_qkv, + RoPE on q/k, write Q,K [B,H,S,DH], V^T [B,H,DH,S]
// One wave computes a 32(M) x 64(N) tile (2 A-frags x 4 B-frags = 8 WMMA per
// 32-deep k-step, 12 b128 loads -> 1.5 loads/WMMA). 64-wide N tiles are
// head-aligned so RoPE's rotate-half partner (d^32) is accumulator a^2.
// ---------------------------------------------------------------------------
__global__ void qkv_rope_kernel(const float* __restrict__ freqs,
                                const __bf16* __restrict__ xb,     // [4096][1024]
                                const __bf16* __restrict__ wT,     // [3072][1024]
                                __bf16* __restrict__ qw,           // [B,H,S,DH]
                                __bf16* __restrict__ kw,           // [B,H,S,DH]
                                __bf16* __restrict__ vT) {         // [B,H,DH,S]
  int lane = threadIdx.x & 31, wave = threadIdx.x >> 5;
  int half_ = lane >> 4, lane16 = lane & 15;
  int nTile = blockIdx.x;                 // 0..47
  int mTile = blockIdx.y * 4 + wave;      // 0..127
  int m0 = mTile * 32, j0 = nTile * 64;

  v8f acc[2][4] = {{v8f{}, v8f{}, v8f{}, v8f{}}, {v8f{}, v8f{}, v8f{}, v8f{}}};
  for (int kk = 0; kk < D_; kk += 32) {
    v16bf af0 = load_a_frag(xb, m0 + lane16, D_, kk, half_);
    v16bf af1 = load_a_frag(xb, m0 + 16 + lane16, D_, kk, half_);
#pragma unroll
    for (int a = 0; a < 4; ++a) {
      v16bf bf_ = load_b_frag(wT, j0 + a * 16 + lane16, D_, kk, half_);
      acc[0][a] = wmma_bf16(af0, bf_, acc[0][a]);
      acc[1][a] = wmma_bf16(af1, bf_, acc[1][a]);
    }
  }

  int part = j0 >> 10;          // 0=q 1=k 2=v
  int h    = (j0 & 1023) >> 6;  // head (tile is head-aligned)
#pragma unroll
  for (int mi = 0; mi < 2; ++mi) {
#pragma unroll
    for (int a = 0; a < 4; ++a) {
#pragma unroll
      for (int r = 0; r < 8; ++r) {
        int mg = m0 + mi * 16 + r + 8 * half_;
        int b  = mg >> 11, s = mg & (S_ - 1);
        int d  = a * 16 + lane16;
        size_t bh = (size_t)b * H_ + h;
        float v = acc[mi][a][r];
        if (part == 2) {
          vT[(bh * DH_ + d) * S_ + s] = (__bf16)v;
        } else {
          // rotate_half partner d^32 lives in accumulator a^2, same lane/vgpr
          float pv = (a < 2) ? -acc[mi][a + 2][r] : acc[mi][a - 2][r];
          float f = freqs[(size_t)s * DH_ + d];
          float sn, cs;
          __sincosf(f, &sn, &cs);
          float ro = v * cs + pv * sn;
          __bf16* dst = (part == 0) ? qw : kw;
          dst[(bh * S_ + s) * DH_ + d] = (__bf16)ro;
        }
      }
    }
  }
}

// ---------------------------------------------------------------------------
// Stage 2: causal flash attention. One wave = one 16-row Q tile of one (b,h).
// Streams KV in 32-row chunks: 4 WMMA scores + online softmax + 4 WMMA P@V.
// ---------------------------------------------------------------------------
__global__ void attn_kernel(const __bf16* __restrict__ qw,
                            const __bf16* __restrict__ kw,
                            const __bf16* __restrict__ vT,
                            __bf16* __restrict__ attnb) {        // [B*S][1024]
  __shared__ __bf16 plds[4 * 16 * 32];     // per-wave P staging (C->A relayout)
  int lane = threadIdx.x & 31, wave = threadIdx.x >> 5;
  int half_ = lane >> 4, lane16 = lane & 15;
  int gw = blockIdx.x * 4 + wave;
  int qt = gw & (S_ / 16 - 1);
  int bh = gw >> 7;
  int q0 = qt * 16;
  const __bf16* qb = qw + (size_t)bh * S_ * DH_;
  const __bf16* kb = kw + (size_t)bh * S_ * DH_;
  const __bf16* vb = vT + (size_t)bh * DH_ * S_;
  __bf16* pl = plds + wave * 512;

  v16bf qf0 = load_a_frag(qb, q0 + lane16, DH_, 0, half_);
  v16bf qf1 = load_a_frag(qb, q0 + lane16, DH_, 32, half_);

  v8f o[4] = {v8f{}, v8f{}, v8f{}, v8f{}};
  float rmax[8], rsum[8];
#pragma unroll
  for (int r = 0; r < 8; ++r) { rmax[r] = -__builtin_inff(); rsum[r] = 0.f; }

  const float scale = 0.125f;  // 1/sqrt(64)
  int nch = (q0 + 15) / 32 + 1;
  for (int kc = 0; kc < nch; ++kc) {
    float p[2][8];
#pragma unroll
    for (int t = 0; t < 2; ++t) {
      int k0 = kc * 32 + t * 16;
      if (k0 <= q0 + 15) {
        v16bf kf0 = load_b_frag(kb, k0 + lane16, DH_, 0, half_);
        v16bf kf1 = load_b_frag(kb, k0 + lane16, DH_, 32, half_);
        v8f s = v8f{};
        s = wmma_bf16(qf0, kf0, s);
        s = wmma_bf16(qf1, kf1, s);
        int kg = k0 + lane16;
#pragma unroll
        for (int r = 0; r < 8; ++r) {
          int mg = q0 + r + 8 * half_;
          p[t][r] = (kg <= mg) ? s[r] * scale : -__builtin_inff();
        }
      } else {
#pragma unroll
        for (int r = 0; r < 8; ++r) p[t][r] = -__builtin_inff();
      }
    }
    // online softmax: row reductions across the 16-lane half (C-layout rows)
    float alpha[8];
#pragma unroll
    for (int r = 0; r < 8; ++r) {
      float cm = fmaxf(p[0][r], p[1][r]);
#pragma unroll
      for (int m = 1; m <= 8; m <<= 1) cm = fmaxf(cm, __shfl_xor(cm, m, 32));
      float nm = fmaxf(rmax[r], cm);
      alpha[r] = __expf(rmax[r] - nm);
      rmax[r] = nm;
    }
#pragma unroll
    for (int t = 0; t < 2; ++t)
#pragma unroll
      for (int r = 0; r < 8; ++r) {
        float e = __expf(p[t][r] - rmax[r]);
        p[t][r] = e;
        pl[(r + 8 * half_) * 32 + t * 16 + lane16] = (__bf16)e;
      }
#pragma unroll
    for (int r = 0; r < 8; ++r) {
      float cs = p[0][r] + p[1][r];
#pragma unroll
      for (int m = 1; m <= 8; m <<= 1) cs += __shfl_xor(cs, m, 32);
      rsum[r] = rsum[r] * alpha[r] + cs;
    }
#pragma unroll
    for (int a = 0; a < 4; ++a)
#pragma unroll
      for (int r = 0; r < 8; ++r) o[a][r] *= alpha[r];

    asm volatile("s_wait_dscnt 0" ::: "memory");   // P stores -> P loads (same wave)
    v16bf pf = load_a_frag(pl, lane16, 32, 0, half_);
#pragma unroll
    for (int c = 0; c < 4; ++c) {
      v16bf vf = load_b_frag(vb, c * 16 + lane16, S_, kc * 32, half_);
      o[c] = wmma_bf16(pf, vf, o[c]);
    }
  }

  int b = bh >> 4, h = bh & 15;
#pragma unroll
  for (int r = 0; r < 8; ++r) {
    float inv = 1.0f / rsum[r];
    int s = q0 + r + 8 * half_;
#pragma unroll
    for (int a = 0; a < 4; ++a) {
      int d = a * 16 + lane16;
      attnb[((size_t)b * S_ + s) * D_ + h * DH_ + d] = (__bf16)(o[a][r] * inv);
    }
  }
}

// ---------------------------------------------------------------------------
// Stage 3: out = attn @ w_out  (fp32 output). 32x64 wave tile like stage 1.
// ---------------------------------------------------------------------------
__global__ void out_gemm_kernel(const __bf16* __restrict__ attnb,  // [4096][1024]
                                const __bf16* __restrict__ woT,    // [1024][1024]
                                float* __restrict__ out) {
  int lane = threadIdx.x & 31, wave = threadIdx.x >> 5;
  int half_ = lane >> 4, lane16 = lane & 15;
  int nTile = blockIdx.x;             // 0..15
  int mTile = blockIdx.y * 4 + wave;  // 0..127
  int m0 = mTile * 32, j0 = nTile * 64;

  v8f acc[2][4] = {{v8f{}, v8f{}, v8f{}, v8f{}}, {v8f{}, v8f{}, v8f{}, v8f{}}};
  for (int kk = 0; kk < D_; kk += 32) {
    v16bf af0 = load_a_frag(attnb, m0 + lane16, D_, kk, half_);
    v16bf af1 = load_a_frag(attnb, m0 + 16 + lane16, D_, kk, half_);
#pragma unroll
    for (int a = 0; a < 4; ++a) {
      v16bf bf_ = load_b_frag(woT, j0 + a * 16 + lane16, D_, kk, half_);
      acc[0][a] = wmma_bf16(af0, bf_, acc[0][a]);
      acc[1][a] = wmma_bf16(af1, bf_, acc[1][a]);
    }
  }
#pragma unroll
  for (int mi = 0; mi < 2; ++mi)
#pragma unroll
    for (int a = 0; a < 4; ++a)
#pragma unroll
      for (int r = 0; r < 8; ++r)
        out[(size_t)(m0 + mi * 16 + r + 8 * half_) * D_ + j0 + a * 16 + lane16] =
            acc[mi][a][r];
}

// ---------------------------------------------------------------------------
// Launcher
// ---------------------------------------------------------------------------
extern "C" void kernel_launch(void* const* d_in, const int* in_sizes, int n_in,
                              void* d_out, int out_size, void* d_ws, size_t ws_size,
                              hipStream_t stream) {
  const float* x      = (const float*)d_in[0];  // [2,2048,1024]
  const float* w_qkv  = (const float*)d_in[1];  // [1024,3072]
  const float* w_out  = (const float*)d_in[2];  // [1024,1024]
  const float* freqs  = (const float*)d_in[3];  // [2048,64]
  float* out = (float*)d_out;

  const int M = B_ * S_;              // 4096
  char* ws = (char*)d_ws;
  size_t off = 0;
  __bf16* xb    = (__bf16*)(ws + off); off += (size_t)M * D_ * 2;        // 8 MB
  __bf16* wqT   = (__bf16*)(ws + off); off += (size_t)3 * D_ * D_ * 2;   // 6 MB
  __bf16* woT   = (__bf16*)(ws + off); off += (size_t)D_ * D_ * 2;       // 2 MB
  __bf16* qw    = (__bf16*)(ws + off); off += (size_t)M * D_ * 2;        // 8 MB
  __bf16* kw    = (__bf16*)(ws + off); off += (size_t)M * D_ * 2;        // 8 MB
  __bf16* vT    = (__bf16*)(ws + off); off += (size_t)M * D_ * 2;        // 8 MB
  __bf16* attnb = (__bf16*)(ws + off); off += (size_t)M * D_ * 2;        // 8 MB
  (void)ws_size; (void)in_sizes; (void)n_in; (void)out_size;

  // fp32 -> bf16 conversions / weight transposes (k-contiguous B layout)
  f32_to_bf16_kernel<<<(M * D_ / 4 + 255) / 256, 256, 0, stream>>>(x, xb, M * D_ / 4);
  transpose_to_bf16_kernel<<<dim3(3 * D_ / 32, D_ / 32), dim3(32, 8), 0, stream>>>(
      w_qkv, wqT, D_, 3 * D_);
  transpose_to_bf16_kernel<<<dim3(D_ / 32, D_ / 32), dim3(32, 8), 0, stream>>>(
      w_out, woT, D_, D_);

  // Stage 1: QKV GEMM + RoPE (128 mTiles x 48 nTiles, 4 waves/block)
  qkv_rope_kernel<<<dim3(48, 32), 128, 0, stream>>>(freqs, xb, wqT, qw, kw, vT);

  // Stage 2: flash attention (B*H*(S/16) = 4096 waves, 4 waves/block)
  attn_kernel<<<1024, 128, 0, stream>>>(qw, kw, vT, attnb);

  // Stage 3: output GEMM (128 mTiles x 16 nTiles, 4 waves/block)
  out_gemm_kernel<<<dim3(16, 32), 128, 0, stream>>>(attnb, woT, out);
}
